// MofNet_59304908423185
// MI455X (gfx1250) — matrix operand order repaired and checked by reading, not compiled, not run on hardware
//
#include <hip/hip_runtime.h>

#define BB   64
#define NNN  512
#define LLN  12
#define DD   128
#define ZZ   384
#define NCV  3
#define MT   4            // M-tiles (nodes) per block: amortizes B-fragment loads
#define BN_EPS 1e-3f

typedef __attribute__((ext_vector_type(16))) _Float16 v16h;
typedef __attribute__((ext_vector_type(8)))  _Float16 v8h;
typedef __attribute__((ext_vector_type(8)))  float    v8f;

// ---------------------------------------------------------------------------
// v0 = embed(node_w, node_types) + slot_types        (B,N,D)
// ---------------------------------------------------------------------------
__global__ __launch_bounds__(256)
void embed_kernel(const int* __restrict__ node_types,
                  const float* __restrict__ slot,
                  const float* __restrict__ node_w,
                  float* __restrict__ v0)
{
    int idx = blockIdx.x * 256 + threadIdx.x;           // over B*N*D
    if (idx >= BB * NNN * DD) return;
    int d  = idx % DD;
    int bn = idx / DD;
    int nt = node_types[bn];
    float emb = (nt >= 0) ? node_w[(size_t)nt * DD + d] : 0.f;
    v0[idx] = emb + slot[idx];
}

// ---------------------------------------------------------------------------
// fp32 -> f16 weight conversion (wf and ws, NC*Z*D each)
// ---------------------------------------------------------------------------
__global__ __launch_bounds__(256)
void cvt_kernel(const float* __restrict__ wf, const float* __restrict__ wsm,
                _Float16* __restrict__ wf16, _Float16* __restrict__ ws16)
{
    int idx = blockIdx.x * 256 + threadIdx.x;
    if (idx >= NCV * ZZ * DD) return;
    wf16[idx] = (_Float16)wf[idx];
    ws16[idx] = (_Float16)wsm[idx];
}

// ---------------------------------------------------------------------------
// One graph-conv layer, fused: z-gather -> WMMA GEMMs -> gate -> neighbor
// sum -> residual -> validity -> masked BatchNorm.
// Block = 256 threads (8 wave32), one block per MT consecutive (b, n) pairs.
// Each wave owns a 16-wide D-column slice; B fragments (wf/ws K-panels) are
// loaded once per K-step and reused across the MT A-tiles (4x less L2 weight
// traffic per WMMA); 8 WMMAs per wave per K-step, 96 per wave per block.
// ---------------------------------------------------------------------------
__global__ __launch_bounds__(256)
void gate_kernel(const float* __restrict__ vin, float* __restrict__ vout,
                 const int* __restrict__ nlist, const int* __restrict__ etypes,
                 const float* __restrict__ edge_w,
                 const _Float16* __restrict__ wf16, const _Float16* __restrict__ ws16,
                 const float* __restrict__ bf, const float* __restrict__ bs,
                 const float* __restrict__ gma, const float* __restrict__ bta,
                 const float* __restrict__ mu, const float* __restrict__ var,
                 int conv)
{
    // MT tiles of 16 rows x 384 halves; row stride 392 halves (784B,
    // 16B aligned, banks rotate by 4 per row -> conflict-free b128 reads)
    __shared__ __align__(16) _Float16 zt[MT][16][392];
    __shared__ float vrow[MT][DD];
    __shared__ float nm[MT][16];
    __shared__ int   validFlag[MT];

    const int bn0  = blockIdx.x * MT;      // MT nodes, same b (N % MT == 0)
    const int b    = bn0 / NNN;
    const int tid  = threadIdx.x;
    const int lane = tid & 31;
    const int wave = tid >> 5;

    if (tid < MT) validFlag[tid] = 1;
    if (tid < MT * 16) {
        int t = tid >> 4, r = tid & 15;
        nm[t][r] = (r < LLN && nlist[(size_t)(bn0 + t) * LLN + r] >= 0) ? 1.f : 0.f;
    }

    // ---- stage z = [v(b,n) | mask*v(b,nbr) | embed(edge)] as f16 ------------
    for (int idx = tid; idx < MT * 16 * ZZ; idx += 256) {
        int t   = idx / (16 * ZZ);
        int rem = idx - t * (16 * ZZ);
        int row = rem / ZZ;
        int c   = rem % ZZ;
        int bn  = bn0 + t;
        float val = 0.f;
        if (row < LLN) {
            if (c < DD) {
                val = vin[(size_t)bn * DD + c];
            } else if (c < 2 * DD) {
                int nl = nlist[(size_t)bn * LLN + row];
                if (nl >= 0) val = vin[((size_t)b * NNN + nl) * DD + (c - DD)];
            } else {
                int et = etypes[(size_t)bn * LLN + row];
                if (et >= 0) val = edge_w[(size_t)et * DD + (c - 2 * DD)];
            }
        }
        zt[t][row][c] = (_Float16)val;
    }
    __syncthreads();

    // ---- WMMA accumulation: accF[t] = z_t@wf, accS[t] = z_t@ws --------------
    const int ct   = wave * 16;          // this wave's 16-wide D-column slice
    const int mrow = lane & 15;          // A: M = lane%16
    const int khi  = (lane >> 4) * 8;    // A: lanes>=16 shift K by +8
    const _Float16* wfbase = wf16 + (size_t)conv * ZZ * DD + ct;
    const _Float16* wsbase = ws16 + (size_t)conv * ZZ * DD + ct;

    v8f accF[MT] = {};
    v8f accS[MT] = {};
    for (int k0 = 0; k0 < ZZ; k0 += 32) {
        // B fragments: lane holds N column (lane%16), halves 0..7 <- K=k0+khi+0..7, 8..15 <- +16.
        // Loaded once, reused across the MT A-tiles.
        const _Float16* wp = wfbase + (size_t)(k0 + khi) * DD + mrow;
        const _Float16* sp = wsbase + (size_t)(k0 + khi) * DD + mrow;
        v16h bF, bS;
#pragma unroll
        for (int s = 0; s < 16; ++s) {
            int kk = (s < 8) ? s : s + 8;
            bF[s] = wp[(size_t)kk * DD];
            bS[s] = sp[(size_t)kk * DD];
        }

#pragma unroll
        for (int t = 0; t < MT; ++t) {
            // A fragment: halves 0..7 <- K=k0+khi+0..7, halves 8..15 <- +16
            const _Float16* rowp = &zt[t][mrow][k0 + khi];
            v8h alo = *(const v8h*)(rowp);
            v8h ahi = *(const v8h*)(rowp + 16);
            v16h a  = __builtin_shufflevector(alo, ahi,
                          0,1,2,3,4,5,6,7,8,9,10,11,12,13,14,15);

            accF[t] = __builtin_amdgcn_wmma_f32_16x16x32_f16(
                          false, a, false, bF, (short)0, accF[t], false, false);
            accS[t] = __builtin_amdgcn_wmma_f32_16x16x32_f16(
                          false, a, false, bS, (short)0, accS[t], false, false);
        }
    }

    // ---- gate + masked sum over neighbor rows -------------------------------
    // C layout: VGPR j -> M=j (lanes 0-15) / M=j+8 (lanes 16-31), N=lane%16
    const int d     = ct + (lane & 15);
    const float bfd = bf[conv * DD + d];
    const float bsd = bs[conv * DD + d];
    const int mbase = (lane >> 4) * 8;
#pragma unroll
    for (int t = 0; t < MT; ++t) {
        float psum = 0.f;
#pragma unroll
        for (int j = 0; j < 8; ++j) {
            float g = 1.f / (1.f + expf(-(accF[t][j] + bfd)));
            float h = tanhf(accS[t][j] + bsd);
            psum += g * h * nm[t][mbase + j];
        }
        psum += __shfl_down(psum, 16, 32);  // combine M=0..7 and M=8..15 halves
        if (lane < 16) vrow[t][d] = vin[(size_t)(bn0 + t) * DD + d] + psum;
    }
    __syncthreads();

    // ---- validity: all(|v_next| > 1e-5) over D ------------------------------
    if (tid < DD) {
#pragma unroll
        for (int t = 0; t < MT; ++t)
            if (fabsf(vrow[t][tid]) <= 1e-5f) atomicAnd(&validFlag[t], 0);
    }
    __syncthreads();

    // ---- masked inference BatchNorm -----------------------------------------
    if (tid < DD) {
        float g  = gma[conv * DD + tid];
        float be = bta[conv * DD + tid];
        float m  = mu[conv * DD + tid];
        float rs = rsqrtf(var[conv * DD + tid] + BN_EPS);
#pragma unroll
        for (int t = 0; t < MT; ++t) {
            float bv = g * (vrow[t][tid] - m) * rs + be;
            vout[(size_t)(bn0 + t) * DD + tid] = validFlag[t] ? bv : 0.f;
        }
    }
}

// ---------------------------------------------------------------------------
// masked mean pool over nodes + dense (D -> 1)
// ---------------------------------------------------------------------------
__global__ __launch_bounds__(128)
void pool_kernel(const float* __restrict__ v, const int* __restrict__ node_types,
                 const float* __restrict__ dense_w, const float* __restrict__ dense_b,
                 float* __restrict__ out)
{
    __shared__ float part[DD];
    const int b = blockIdx.x;
    const int d = threadIdx.x;
    float sum = 0.f, cnt = 0.f;
    for (int n = 0; n < NNN; ++n) {
        float pm = (node_types[(size_t)b * NNN + n] >= 0) ? 1.f : 0.f;
        sum += v[((size_t)b * NNN + n) * DD + d] * pm;
        cnt += pm;
    }
    part[d] = (sum / cnt) * dense_w[d];
    __syncthreads();
    for (int s = 64; s > 0; s >>= 1) {
        if (d < s) part[d] += part[d + s];
        __syncthreads();
    }
    if (d == 0) out[b] = part[0] + dense_b[0];
}

// ---------------------------------------------------------------------------
extern "C" void kernel_launch(void* const* d_in, const int* in_sizes, int n_in,
                              void* d_out, int out_size, void* d_ws, size_t ws_size,
                              hipStream_t stream)
{
    (void)in_sizes; (void)n_in; (void)out_size; (void)ws_size;

    const int*   node_types = (const int*)  d_in[0];
    const int*   nlist      = (const int*)  d_in[1];
    const int*   etypes     = (const int*)  d_in[2];
    const float* slot       = (const float*)d_in[3];
    const float* node_w     = (const float*)d_in[4];
    const float* edge_w     = (const float*)d_in[5];
    const float* wf         = (const float*)d_in[6];
    const float* bf         = (const float*)d_in[7];
    const float* wsm        = (const float*)d_in[8];
    const float* bs         = (const float*)d_in[9];
    const float* gma        = (const float*)d_in[10];
    const float* bta        = (const float*)d_in[11];
    const float* mu         = (const float*)d_in[12];
    const float* var        = (const float*)d_in[13];
    const float* dense_w    = (const float*)d_in[14];
    const float* dense_b    = (const float*)d_in[15];

    char* wsb = (char*)d_ws;
    const size_t vbytes = (size_t)BB * NNN * DD * sizeof(float);       // 16 MB
    const size_t wbytes = ((size_t)NCV * ZZ * DD * sizeof(_Float16) + 255) & ~(size_t)255;
    float*    vA   = (float*)   (wsb);
    float*    vB   = (float*)   (wsb + vbytes);
    _Float16* wf16 = (_Float16*)(wsb + 2 * vbytes);
    _Float16* ws16 = (_Float16*)(wsb + 2 * vbytes + wbytes);

    embed_kernel<<<(BB * NNN * DD + 255) / 256, 256, 0, stream>>>(node_types, slot, node_w, vA);
    cvt_kernel<<<(NCV * ZZ * DD + 255) / 256, 256, 0, stream>>>(wf, wsm, wf16, ws16);

    float* vin  = vA;
    float* vout = vB;
    for (int c = 0; c < NCV; ++c) {
        gate_kernel<<<BB * NNN / MT, 256, 0, stream>>>(vin, vout, nlist, etypes, edge_w,
                                                       wf16, ws16, bf, bs,
                                                       gma, bta, mu, var, c);
        float* t = vin; vin = vout; vout = t;
    }

    pool_kernel<<<BB, 128, 0, stream>>>(vin, node_types, dense_w, dense_b, (float*)d_out);
}
